// LightGlueMOT_17248588660910
// MI455X (gfx1250) — matrix operand order repaired and checked by reading, compiled, not run on hardware
//
#include <hip/hip_runtime.h>
#include <hip/hip_bf16.h>

typedef __attribute__((ext_vector_type(16))) _Float16 v16h;
typedef __attribute__((ext_vector_type(8)))  _Float16 v8h;
typedef __attribute__((ext_vector_type(8)))  float    v8f;

#define TPB 256
#define NEG_SLOPE 0.2f
#define NT 4   // n-tiles (16 cols each) per wave

__device__ __forceinline__ void atomicMaxFloat(float* addr, float val) {
    int* ia = (int*)addr;
    int old = __float_as_int(*addr);
    while (__int_as_float(old) < val) {
        int assumed = old;
        old = atomicCAS(ia, assumed, __float_as_int(val));
        if (old == assumed) break;
    }
}

// ---------------- elementwise / staging kernels ----------------

__global__ void k_fill_f32(float* __restrict__ p, float v, long n) {
    long i = (long)blockIdx.x * blockDim.x + threadIdx.x;
    if (i < n) p[i] = v;
}

__global__ void k_cvt(const float* __restrict__ in, _Float16* __restrict__ out, long n) {
    long i = (long)blockIdx.x * blockDim.x + threadIdx.x;
    if (i < n) out[i] = (_Float16)in[i];
}

// Pack weight W[O,I] (row-major, O = out features = GEMM Nc, I = K) into
// WMMA B-fragment order: Bp[((kt*tilesN + nt)*32 + lane)*16 + e]
//   = W[nt*16 + (lane&15)][kt*32 + (lane>>4)*16 + e]
// so each lane's 16-element fragment is one contiguous 32-byte run.
__global__ void k_pack_b(const float* __restrict__ W, _Float16* __restrict__ Bp,
                         int O, int I) {
    long idx = (long)blockIdx.x * blockDim.x + threadIdx.x;
    long total = (long)O * I;
    if (idx >= total) return;
    int  e    = (int)(idx & 15);
    int  lane = (int)((idx >> 4) & 31);
    long blk  = idx >> 9;                 // kt*tilesN + nt
    int  tilesN = O >> 4;
    int  kt = (int)(blk / tilesN);
    int  nt = (int)(blk % tilesN);
    int  k = kt * 32 + (lane >> 4) * 16 + e;
    int  o = nt * 16 + (lane & 15);
    Bp[idx] = (_Float16)W[(long)o * I + k];
}

// ---------------- WMMA GEMM: C[M,Nc] = act(A[M,K] @ W^T + bias) ----------------
// one wave per 16x(16*NT) output strip; f16 in, f32 accumulate.
// A fragments: two b128 loads per k-step (contiguous k = k0+16h+s, same order as packed B);
// B fragments: pre-packed, two b128 loads per tile, coalesced across lanes.
__global__ void k_gemm_wmma(const _Float16* __restrict__ A, const _Float16* __restrict__ Bp,
                            const float* __restrict__ bias, float* __restrict__ Cf,
                            _Float16* __restrict__ Ch, int M, int Nc, int K, int act) {
    long gid  = (long)blockIdx.x * blockDim.x + threadIdx.x;
    int  wave = (int)(gid >> 5);
    int  lane = (int)(gid & 31);
    int  tilesN  = Nc >> 4;
    int  groupsN = tilesN / NT;
    int  tm = (wave / groupsN) << 4;       // wave-uniform
    int  tg = (wave % groupsN) * NT;       // first n-tile of this wave's strip
    if (tm >= M) return;                   // wave-uniform -> EXEC all-1s at WMMA

    int m  = lane & 15;
    int hh = lane >> 4;

    v8f acc[NT];
#pragma unroll
    for (int t = 0; t < NT; t++) acc[t] = (v8f){0.f,0.f,0.f,0.f,0.f,0.f,0.f,0.f};

    const _Float16* __restrict__ arow = A + (long)(tm + m) * K;

    union Frag { v16h v; v8h h[2]; };

    int kt = 0;
    for (int k0 = 0; k0 < K; k0 += 32, kt++) {
        Frag af;
        af.h[0] = *(const v8h*)(arow + k0 + hh * 16);       // K = k0+16h .. +7
        af.h[1] = *(const v8h*)(arow + k0 + hh * 16 + 8);   // K = k0+16h+8 .. +15
        const _Float16* __restrict__ bbase =
            Bp + (((long)kt * tilesN + tg) * 32 + lane) * 16;
#pragma unroll
        for (int t = 0; t < NT; t++) {
            Frag bf;
            bf.h[0] = *(const v8h*)(bbase + (long)t * 512);
            bf.h[1] = *(const v8h*)(bbase + (long)t * 512 + 8);
            acc[t] = __builtin_amdgcn_wmma_f32_16x16x32_f16(
                false, af.v, false, bf.v, (short)0, acc[t], false, false);
        }
    }

#pragma unroll
    for (int t = 0; t < NT; t++) {
        int col = (tg + t) * 16 + m;
        float bv = bias ? bias[col] : 0.0f;
#pragma unroll
        for (int r = 0; r < 8; r++) {
            int row = tm + r + 8 * hh;     // D layout: VGPR r -> M = r + 8*(lane>=16)
            float v = acc[t][r] + bv;
            if (act == 1) v = fmaxf(v, 0.0f);
            long o = (long)row * Nc + col;
            if (Cf) Cf[o] = v;
            if (Ch) Ch[o] = (_Float16)v;
        }
    }
}

// ---------------- GATv2 attention kernels ----------------

__global__ void k_logits(const float* __restrict__ xl, const float* __restrict__ att,
                         const int* __restrict__ src, const int* __restrict__ dst,
                         float* __restrict__ logits, int E_) {
    long idx = (long)blockIdx.x * blockDim.x + threadIdx.x;
    if (idx >= (long)E_ * 4) return;
    int e = (int)(idx >> 2), h = (int)(idx & 3);
    const float* a  = att + h * 64;
    const float* xs = xl + (long)src[e] * 256 + h * 64;
    const float* xd = xl + (long)dst[e] * 256 + h * 64;
    float acc = 0.f;
#pragma unroll 8
    for (int c = 0; c < 64; c++) {
        float v = xd[c] - xs[c];
        v = (v > 0.f) ? v : NEG_SLOPE * v;
        acc += v * a[c];
    }
    logits[idx] = acc;
}

__global__ void k_segmax(const float* __restrict__ logits, const int* __restrict__ dst,
                         float* __restrict__ mbuf, int E_) {
    long idx = (long)blockIdx.x * blockDim.x + threadIdx.x;
    if (idx >= (long)E_ * 4) return;
    int e = (int)(idx >> 2), h = (int)(idx & 3);
    atomicMaxFloat(&mbuf[(long)dst[e] * 4 + h], logits[idx]);
}

__global__ void k_expdenom(float* __restrict__ logits, const int* __restrict__ dst,
                           const float* __restrict__ mbuf, float* __restrict__ denom, int E_) {
    long idx = (long)blockIdx.x * blockDim.x + threadIdx.x;
    if (idx >= (long)E_ * 4) return;
    int e = (int)(idx >> 2), h = (int)(idx & 3);
    float ex = __expf(logits[idx] - mbuf[(long)dst[e] * 4 + h]);
    logits[idx] = ex;
    atomicAdd(&denom[(long)dst[e] * 4 + h], ex);
}

__global__ void k_agg(const float* __restrict__ ea, const float* __restrict__ ex,
                      const float* __restrict__ denom, const int* __restrict__ dst,
                      float* __restrict__ agg, int E_) {
    long idx = (long)blockIdx.x * blockDim.x + threadIdx.x;
    if (idx >= (long)E_ * 256) return;
    int e = (int)(idx >> 8), j = (int)(idx & 255), h = j >> 6;
    int d = dst[e];
    float alpha = ex[(long)e * 4 + h] / (denom[(long)d * 4 + h] + 1e-16f);
    atomicAdd(&agg[(long)d * 256 + j], ea[idx] * alpha);
}

__global__ void k_cat(const float* __restrict__ xl, const float* __restrict__ agg,
                      _Float16* __restrict__ cat, int Nn) {
    long idx = (long)blockIdx.x * blockDim.x + threadIdx.x;
    if (idx >= (long)Nn * 4 * 128) return;
    int row = (int)(idx >> 7), j = (int)(idx & 127);
    int n = row >> 2, h = row & 3;
    float v = (j < 64) ? xl[(long)n * 256 + h * 64 + j]
                       : agg[(long)n * 256 + h * 64 + (j - 64)];
    cat[idx] = (_Float16)v;
}

__global__ void k_node(const float* __restrict__ upd, const float* __restrict__ gbias,
                       float* __restrict__ nodeF32, _Float16* __restrict__ nodeF16, int Nn) {
    long idx = (long)blockIdx.x * blockDim.x + threadIdx.x;
    if (idx >= (long)Nn * 256) return;
    float v = upd[idx] + gbias[idx & 255];
    nodeF32[idx] = v;
    nodeF16[idx] = (_Float16)v;
}

__global__ void k_tmp(const float* __restrict__ edgeF32, const float* __restrict__ nodeF32,
                      const int* __restrict__ src, const int* __restrict__ dst,
                      _Float16* __restrict__ tmp, int E_) {
    long idx = (long)blockIdx.x * blockDim.x + threadIdx.x;
    if (idx >= (long)E_ * 512) return;
    int e = (int)(idx >> 9), j = (int)(idx & 511);
    float v;
    if (j < 256) v = edgeF32[(long)e * 256 + j];
    else {
        int jj = j - 256;
        v = nodeF32[(long)src[e] * 256 + jj] - nodeF32[(long)dst[e] * 256 + jj];
    }
    tmp[idx] = (_Float16)v;
}

// ---------------- final scoring kernels ----------------

__global__ void k_vals(const float* __restrict__ node, const int* __restrict__ src,
                       const int* __restrict__ dst, float* __restrict__ vals, int E_) {
    int e = blockIdx.x * blockDim.x + threadIdx.x;
    if (e >= E_) return;
    const float* a = node + (long)src[e] * 256;
    const float* b = node + (long)dst[e] * 256;
    float acc = 0.f;
#pragma unroll 8
    for (int c = 0; c < 256; c++) acc += a[c] * b[c];
    vals[e] = acc * 0.0625f;   // / sqrt(256)
}

__global__ void k_scatter(float* __restrict__ sim, const int* __restrict__ src,
                          const int* __restrict__ dst, const float* __restrict__ vals,
                          int E_, int Nn) {
    int e = blockIdx.x * blockDim.x + threadIdx.x;
    if (e >= E_) return;
    sim[(long)src[e] * Nn + dst[e]] = vals[e];
}

__global__ void k_rowstats(const float* __restrict__ sim, float* __restrict__ rmax,
                           float* __restrict__ rsum, int n) {
    __shared__ float red[TPB];
    int row = blockIdx.x, t = threadIdx.x;
    const float* p = sim + (long)row * n;
    float m = -3.402823466e+38f;
    for (int j = t; j < n; j += TPB) m = fmaxf(m, p[j]);
    red[t] = m; __syncthreads();
    for (int s = TPB / 2; s > 0; s >>= 1) { if (t < s) red[t] = fmaxf(red[t], red[t + s]); __syncthreads(); }
    float rm = red[0]; __syncthreads();
    float sum = 0.f;
    for (int j = t; j < n; j += TPB) sum += __expf(p[j] - rm);
    red[t] = sum; __syncthreads();
    for (int s = TPB / 2; s > 0; s >>= 1) { if (t < s) red[t] += red[t + s]; __syncthreads(); }
    if (t == 0) { rmax[row] = rm; rsum[row] = red[0]; }
}

__global__ void k_colstats(const float* __restrict__ sim, float* __restrict__ cmax,
                           float* __restrict__ csum, int n) {
    int j = blockIdx.x * blockDim.x + threadIdx.x;
    if (j >= n) return;
    float m = -3.402823466e+38f;
    for (int i = 0; i < n; i++) m = fmaxf(m, sim[(long)i * n + j]);
    float s = 0.f;
    for (int i = 0; i < n; i++) s += __expf(sim[(long)i * n + j] - m);
    cmax[j] = m; csum[j] = s;
}

__global__ void k_combine(float* __restrict__ out, const float* __restrict__ rmax,
                          const float* __restrict__ rsum, const float* __restrict__ cmax,
                          const float* __restrict__ csum, int n) {
    long idx = (long)blockIdx.x * blockDim.x + threadIdx.x;
    if (idx >= (long)n * n) return;
    int i = (int)(idx / n), j = (int)(idx % n);
    float s = out[idx];
    out[idx] = 0.5f * (__expf(s - rmax[i]) / rsum[i] + __expf(s - cmax[j]) / csum[j]);
}

// ---------------- host orchestration ----------------

static inline dim3 gblk(long nthreads) { return dim3((unsigned)((nthreads + TPB - 1) / TPB)); }

extern "C" void kernel_launch(void* const* d_in, const int* in_sizes, int n_in,
                              void* d_out, int out_size, void* d_ws, size_t ws_size,
                              hipStream_t stream) {
    const int Dd = 256, Hh = 4;
    const int Nn = in_sizes[0] / Dd;          // 4096
    const int Ee = in_sizes[2] / 2;           // 131072
    const int Ll = in_sizes[3] / (Dd * Dd);   // 2

    const float* x  = (const float*)d_in[0];
    const float* e0 = (const float*)d_in[1];
    const int*   src = (const int*)d_in[2];
    const int*   dst = src + Ee;
    const float* Wl = (const float*)d_in[3];
    const float* bl = (const float*)d_in[4];
    const float* We = (const float*)d_in[5];
    const float* be = (const float*)d_in[6];
    const float* att = (const float*)d_in[7];
    const float* gbias = (const float*)d_in[8];
    const float* Wp = (const float*)d_in[9];
    const float* bp = (const float*)d_in[10];
    const float* W1 = (const float*)d_in[11];
    const float* b1 = (const float*)d_in[12];
    const float* W2 = (const float*)d_in[13];
    const float* b2 = (const float*)d_in[14];

    // workspace carve-up
    char* p = (char*)d_ws;
    auto alloc = [&](size_t bytes) -> void* {
        void* r = (void*)p; p += (bytes + 255) & ~(size_t)255; return r;
    };
    _Float16* eh      = (_Float16*)alloc((size_t)Ee * 256 * 2);  // edge f16
    void*     tmpU    = alloc((size_t)Ee * 512 * 2);             // union: ea f32 [E,256] / tmp f16 [E,512]
    _Float16* h1h     = (_Float16*)alloc((size_t)Ee * 256 * 2);
    float*    edgeF32 = (float*)alloc((size_t)Ee * 256 * 4);
    _Float16* xh      = (_Float16*)alloc((size_t)Nn * 256 * 2);  // node f16
    float*    xl      = (float*)alloc((size_t)Nn * 256 * 4);     // xl / later nodeF32
    float*    agg     = (float*)alloc((size_t)Nn * 256 * 4);     // agg / later upd
    _Float16* cat     = (_Float16*)alloc((size_t)Nn * Hh * 128 * 2);
    float*    logits  = (float*)alloc((size_t)Ee * Hh * 4);      // logits -> ex in place
    float*    mbuf    = (float*)alloc((size_t)Nn * Hh * 4);
    float*    denom   = (float*)alloc((size_t)Nn * Hh * 4);
    _Float16* wbuf    = (_Float16*)alloc((size_t)512 * 256 * 2); // packed weight staging
    float*    vals    = (float*)alloc((size_t)Ee * 4);
    float*    rmax    = (float*)alloc((size_t)Nn * 4);
    float*    rsum    = (float*)alloc((size_t)Nn * 4);
    float*    cmax    = (float*)alloc((size_t)Nn * 4);
    float*    csum    = (float*)alloc((size_t)Nn * 4);

    float* ea = (float*)tmpU;
    _Float16* tmp = (_Float16*)tmpU;

    auto gemm = [&](const _Float16* A, const _Float16* Bp, const float* bias,
                    float* Cf, _Float16* Ch, int M, int Nc, int K, int act) {
        long waves = (long)(M / 16) * (Nc / 16 / NT);
        k_gemm_wmma<<<gblk(waves * 32), TPB, 0, stream>>>(A, Bp, bias, Cf, Ch, M, Nc, K, act);
    };

    k_cvt<<<gblk((long)Nn * 256), TPB, 0, stream>>>(x, xh, (long)Nn * 256);

    const float* curEdge = e0;
    for (int l = 0; l < Ll; l++) {
        // stage edge features f16
        k_cvt<<<gblk((long)Ee * 256), TPB, 0, stream>>>(curEdge, eh, (long)Ee * 256);

        // xl = node @ Wl^T + bl
        k_pack_b<<<gblk((long)Dd * Dd), TPB, 0, stream>>>(Wl + (long)l * Dd * Dd, wbuf, Dd, Dd);
        gemm(xh, wbuf, bl + (long)l * Dd, xl, nullptr, Nn, Dd, Dd, 0);

        // ea = edge @ We^T + be
        k_pack_b<<<gblk((long)Dd * Dd), TPB, 0, stream>>>(We + (long)l * Dd * Dd, wbuf, Dd, Dd);
        gemm(eh, wbuf, be + (long)l * Dd, ea, nullptr, Ee, Dd, Dd, 0);

        // attention: segment softmax over dst
        k_fill_f32<<<gblk((long)Nn * Hh), TPB, 0, stream>>>(mbuf, -3.402823466e+38f, (long)Nn * Hh);
        k_fill_f32<<<gblk((long)Nn * Hh), TPB, 0, stream>>>(denom, 0.0f, (long)Nn * Hh);
        k_fill_f32<<<gblk((long)Nn * 256), TPB, 0, stream>>>(agg, 0.0f, (long)Nn * 256);
        k_logits<<<gblk((long)Ee * Hh), TPB, 0, stream>>>(xl, att + (long)l * Hh * 64, src, dst, logits, Ee);
        k_segmax<<<gblk((long)Ee * Hh), TPB, 0, stream>>>(logits, dst, mbuf, Ee);
        k_expdenom<<<gblk((long)Ee * Hh), TPB, 0, stream>>>(logits, dst, mbuf, denom, Ee);
        k_agg<<<gblk((long)Ee * 256), TPB, 0, stream>>>(ea, logits, denom, dst, agg, Ee);

        // update: per-head projector [N*H,128] @ Wp^T[128,64] + bp  (upd reuses agg buffer)
        k_cat<<<gblk((long)Nn * Hh * 128), TPB, 0, stream>>>(xl, agg, cat, Nn);
        k_pack_b<<<gblk((long)64 * 128), TPB, 0, stream>>>(Wp + (long)l * 64 * 128, wbuf, 64, 128);
        gemm(cat, wbuf, bp + (long)l * 64, agg, nullptr, Nn * Hh, 64, 128, 0);
        k_node<<<gblk((long)Nn * 256), TPB, 0, stream>>>(agg, gbias + (long)l * Dd, xl, xh, Nn);
        // xl now holds node f32, xh node f16

        // EdgeConv — dead for the final layer (scores use nodes only): skip it
        if (l < Ll - 1) {
            k_tmp<<<gblk((long)Ee * 512), TPB, 0, stream>>>(curEdge, xl, src, dst, tmp, Ee);
            k_pack_b<<<gblk((long)Dd * 2 * Dd), TPB, 0, stream>>>(W1 + (long)l * Dd * 2 * Dd, wbuf, Dd, 2 * Dd);
            gemm(tmp, wbuf, b1 + (long)l * Dd, nullptr, h1h, Ee, Dd, 2 * Dd, 1);   // relu
            k_pack_b<<<gblk((long)Dd * Dd), TPB, 0, stream>>>(W2 + (long)l * Dd * Dd, wbuf, Dd, Dd);
            gemm(h1h, wbuf, b2 + (long)l * Dd, edgeF32, nullptr, Ee, Dd, Dd, 0);
            curEdge = edgeF32;
        }
    }

    // MatchAssignment
    float* out = (float*)d_out;
    k_vals<<<gblk(Ee), TPB, 0, stream>>>(xl, src, dst, vals, Ee);
    k_fill_f32<<<gblk((long)Nn * Nn), TPB, 0, stream>>>(out, -3.40282346638528859811704183485e+38f, (long)Nn * Nn);
    k_scatter<<<gblk(Ee), TPB, 0, stream>>>(out, src, dst, vals, Ee, Nn);
    k_rowstats<<<dim3(Nn), TPB, 0, stream>>>(out, rmax, rsum, Nn);
    k_colstats<<<gblk(Nn), TPB, 0, stream>>>(out, cmax, csum, Nn);
    k_combine<<<gblk((long)Nn * Nn), TPB, 0, stream>>>(out, rmax, rsum, cmax, csum, Nn);
}
